// MultiHeadSelfAttention_5884105195753
// MI455X (gfx1250) — matrix operand order repaired and checked by reading, not compiled, not run on hardware
//
#include <hip/hip_runtime.h>
#include <stdint.h>

// ---------------- problem constants ----------------
#define S_LEN 4096
#define DM    1024
#define NH    16
#define HD    64

// ---------------- WMMA fragment types (wave32, gfx1250) ----------------
typedef __attribute__((ext_vector_type(16))) __bf16 bf16x16;
typedef __attribute__((ext_vector_type(8)))  __bf16 bf16x8;
typedef __attribute__((ext_vector_type(8)))  float  v8f;

// Async global->LDS (CDNA5) if the toolchain exposes the builtin.
#if defined(__HIP_DEVICE_COMPILE__) && defined(__has_builtin)
# if __has_builtin(__builtin_amdgcn_global_load_async_to_lds_b128)
#  define HAVE_ASYNC_LDS 1
# endif
#endif
#ifndef HAVE_ASYNC_LDS
# define HAVE_ASYNC_LDS 0
#endif

// The builtin's pointer params are `int __attribute__((vector_size(16)))*`
// in addrspace(1) (global) / addrspace(3) (LDS).
typedef int i32v4 __attribute__((vector_size(16)));
#define AS1CAST(p) ((__attribute__((address_space(1))) i32v4*)(p))
#define AS3CAST(p) ((__attribute__((address_space(3))) i32v4*)(p))

__device__ __forceinline__ unsigned short f2bf(float f) {
  unsigned int u = __float_as_uint(f);
  u += 0x7FFFu + ((u >> 16) & 1u);          // round-to-nearest-even
  return (unsigned short)(u >> 16);
}

__device__ __forceinline__ v8f wmma_bf16(bf16x16 a, bf16x16 b, v8f c) {
  // D = A(16x32) * B(32x16) + C, f32 accumulate
  return __builtin_amdgcn_wmma_f32_16x16x32_bf16(false, a, false, b, (short)0, c,
                                                 false, false);
}

// A-fragment (16x32 bf16, MxK): lane half=lane/16.
// half 0 holds K {0..15}, half 1 holds K {16..31}  -> two 16B runs.
__device__ __forceinline__ bf16x16 load_a16x32(const unsigned short* rowptr,
                                               int k0, int halfc) {
  bf16x16 a;
  ((bf16x8*)&a)[0] = *(const bf16x8*)(rowptr + k0 + halfc * 16);
  ((bf16x8*)&a)[1] = *(const bf16x8*)(rowptr + k0 + halfc * 16 + 8);
  return a;
}

// ---------------- kernel 0: f32 -> bf16 convert ----------------
__global__ __launch_bounds__(256) void to_bf16_kernel(const float* __restrict__ src,
                                                      unsigned short* __restrict__ dst,
                                                      int n) {
  int i = blockIdx.x * 256 + threadIdx.x;
  if (i < n) dst[i] = f2bf(src[i]);
}

// ---------------- kernel 1: fused QKV GEMM (32x64 per wave) ----------------
// out cols 0..1023 -> qT (f32, [D][S]); 1024..2047 -> kT (f32, [D][S]);
// 2048..3071 -> Vt (bf16, [h*64+d][s])
__global__ __launch_bounds__(128) void qkv_gemm_kernel(
    const unsigned short* __restrict__ xb,
    const unsigned short* __restrict__ Wqb,
    const unsigned short* __restrict__ Wkb,
    const unsigned short* __restrict__ Wvb,
    float* __restrict__ qT, float* __restrict__ kT,
    unsigned short* __restrict__ Vtb) {
  const int wave  = blockIdx.x * 4 + (threadIdx.x >> 5);
  const int lane  = threadIdx.x & 31;
  const int halfc = lane >> 4;
  const int ln    = lane & 15;
  const int mt = wave / 48;          // 128 m-tiles of 32 rows
  const int nb = wave % 48;          // 48 n-blocks of 64 cols (3*1024 total)
  const int m0 = mt * 32;
  const int n0 = nb * 64;
  const int mat = n0 >> 10;          // 0=Q,1=K,2=V
  const int nc0 = n0 & 1023;
  const unsigned short* W = (mat == 0) ? Wqb : (mat == 1) ? Wkb : Wvb;

  v8f acc[2][4];
#pragma unroll
  for (int am = 0; am < 2; ++am)
#pragma unroll
    for (int t = 0; t < 4; ++t)
      acc[am][t] = (v8f){0.f,0.f,0.f,0.f,0.f,0.f,0.f,0.f};

  const unsigned short* arow0 = xb + (size_t)(m0 + ln) * DM;
  const unsigned short* arow1 = xb + (size_t)(m0 + 16 + ln) * DM;
  const unsigned short* brow[4];
#pragma unroll
  for (int t = 0; t < 4; ++t)
    brow[t] = W + (size_t)(nc0 + t * 16 + ln) * DM + halfc * 16;

  // 8 WMMAs per k-step (2 A-frags x 4 B-frags) -> 12 b128 loads / 8 wmma
#pragma unroll 2
  for (int k0 = 0; k0 < DM; k0 += 32) {
    bf16x16 a0 = load_a16x32(arow0, k0, halfc);
    bf16x16 a1 = load_a16x32(arow1, k0, halfc);
    bf16x16 b[4];
#pragma unroll
    for (int t = 0; t < 4; ++t) b[t] = *(const bf16x16*)(brow[t] + k0);
#pragma unroll
    for (int t = 0; t < 4; ++t) acc[0][t] = wmma_bf16(a0, b[t], acc[0][t]);
#pragma unroll
    for (int t = 0; t < 4; ++t) acc[1][t] = wmma_bf16(a1, b[t], acc[1][t]);
  }

  // C/D frag: VGPR i -> row (m0+am*16) + i + 8*halfc, col = n-tile + ln.
  // Transposed store => each lane writes 8 contiguous elements along s.
#pragma unroll
  for (int am = 0; am < 2; ++am) {
    const int srow = m0 + am * 16 + halfc * 8;
    if (mat < 2) {
      float* dstT = (mat == 0) ? qT : kT;
#pragma unroll
      for (int t = 0; t < 4; ++t) {
        float* p = dstT + (size_t)(nc0 + t * 16 + ln) * S_LEN + srow;
        float4 lo = {acc[am][t][0], acc[am][t][1], acc[am][t][2], acc[am][t][3]};
        float4 hi = {acc[am][t][4], acc[am][t][5], acc[am][t][6], acc[am][t][7]};
        *(float4*)(p)     = lo;
        *(float4*)(p + 4) = hi;
      }
    } else {
#pragma unroll
      for (int t = 0; t < 4; ++t) {
        unsigned short* p = Vtb + (size_t)(nc0 + t * 16 + ln) * S_LEN + srow;
        unsigned int w0 = (unsigned)f2bf(acc[am][t][0]) | ((unsigned)f2bf(acc[am][t][1]) << 16);
        unsigned int w1 = (unsigned)f2bf(acc[am][t][2]) | ((unsigned)f2bf(acc[am][t][3]) << 16);
        unsigned int w2 = (unsigned)f2bf(acc[am][t][4]) | ((unsigned)f2bf(acc[am][t][5]) << 16);
        unsigned int w3 = (unsigned)f2bf(acc[am][t][6]) | ((unsigned)f2bf(acc[am][t][7]) << 16);
        uint4 v = {w0, w1, w2, w3};
        *(uint4*)p = v;
      }
    }
  }
}

// ---------------- kernel 2: RoPE + bf16 pack ----------------
__global__ __launch_bounds__(256) void rope_pack_kernel(
    const float* __restrict__ qT, const float* __restrict__ kT,
    const int* __restrict__ tokpos,
    unsigned short* __restrict__ Qb, unsigned short* __restrict__ Kb) {
  __shared__ float tile[64][65];
  const int h  = blockIdx.y;
  const int s0 = blockIdx.x * 64;
  const int which = blockIdx.z;                 // 0 = Q, 1 = K
  const float* src = which ? kT : qT;
  unsigned short* dst = which ? Kb : Qb;
  const float scale = which ? 1.0f : 0.125f;    // 1/sqrt(64) into Q

  for (int idx = threadIdx.x; idx < 64 * 64; idx += 256) {
    int r = idx >> 6, c = idx & 63;             // r = d, c = s-local (coalesced)
    tile[r][c] = src[(size_t)(h * 64 + r) * S_LEN + s0 + c];
  }
  __syncthreads();
  for (int idx = threadIdx.x; idx < 64 * 64; idx += 256) {
    int sl = idx >> 6, d = idx & 63;
    float xv  = tile[d][sl];
    float rot = (d & 1) ? tile[d - 1][sl] : -tile[d + 1][sl];  // (x0,x1)->(-x1,x0)
    int p2 = d & ~1;                                            // 2*pair
    float inv = __powf(10000.0f, -(float)p2 * (1.0f / 64.0f));
    float ang = (float)tokpos[s0 + sl] * inv;
    float sn, cs;
    __sincosf(ang, &sn, &cs);
    dst[((size_t)h * S_LEN + s0 + sl) * HD + d] = f2bf((xv * cs + rot * sn) * scale);
  }
}

// ---------------- kernel 3: causal flash attention ----------------
// block = (head, 64-query block); 4 waves, each owns a 16-row strip.
// K/V tiles staged in LDS once per block (shared by all 4 waves),
// double-buffered with async global->LDS copies when available.

__device__ __forceinline__ void issue_kv_copy(
    const unsigned short* __restrict__ Kh,  // [s][d] bf16
    const unsigned short* __restrict__ Vh,  // [d][s] bf16
    int kbase, unsigned short* kdst, unsigned short* vdst, int tid) {
#pragma unroll
  for (int r = 0; r < 4; ++r) {
    const int c = r * 128 + tid;            // 512 chunks of 16B per tile
    const unsigned short* gk = Kh + (size_t)kbase * HD + (size_t)c * 8; // contiguous
    const int d  = c >> 3;
    const int cc = c & 7;
    const unsigned short* gv = Vh + (size_t)d * S_LEN + kbase + cc * 8;
#if HAVE_ASYNC_LDS
    __builtin_amdgcn_global_load_async_to_lds_b128(AS1CAST(gk), AS3CAST(kdst + c * 8), 0, 0);
    __builtin_amdgcn_global_load_async_to_lds_b128(AS1CAST(gv), AS3CAST(vdst + c * 8), 0, 0);
#else
    *(uint4*)(kdst + c * 8) = *(const uint4*)gk;
    *(uint4*)(vdst + c * 8) = *(const uint4*)gv;
#endif
  }
}

__device__ __forceinline__ void wait_kv_copy() {
#if HAVE_ASYNC_LDS
# if __has_builtin(__builtin_amdgcn_s_wait_asynccnt)
  __builtin_amdgcn_s_wait_asynccnt(0);
# else
  asm volatile("s_wait_asynccnt 0x0" ::: "memory");
# endif
#endif
  __syncthreads();
}

__global__ __launch_bounds__(128) void flash_attn_kernel(
    const unsigned short* __restrict__ Qb,
    const unsigned short* __restrict__ Kb,
    const unsigned short* __restrict__ Vtb,
    unsigned short* __restrict__ ctxb) {
  const int h  = blockIdx.y;
  const int qb = blockIdx.x;
  const int w    = threadIdx.x >> 5;
  const int lane = threadIdx.x & 31;
  const int halfc = lane >> 4;
  const int ln    = lane & 15;
  const int m0 = qb * 64 + w * 16;

  const unsigned short* Qh = Qb  + (size_t)h * S_LEN * HD;
  const unsigned short* Kh = Kb  + (size_t)h * S_LEN * HD;
  const unsigned short* Vh = Vtb + (size_t)h * HD * S_LEN;   // [d][s]

  __shared__ __align__(32) unsigned short kbuf[2][64 * HD];  // 2 x 8KB
  __shared__ __align__(32) unsigned short vbuf[2][64 * HD];  // 2 x 8KB
  __shared__ __align__(32) unsigned short pbuf[4][16][64];   // wave-private P tiles

  bf16x16 qa[2];
  {
    const unsigned short* qrow = Qh + (size_t)(m0 + ln) * HD;
    qa[0] = load_a16x32(qrow, 0, halfc);
    qa[1] = load_a16x32(qrow, 32, halfc);
  }

  v8f ctx[4];
#pragma unroll
  for (int t = 0; t < 4; ++t) ctx[t] = (v8f){0.f,0.f,0.f,0.f,0.f,0.f,0.f,0.f};
  float mrow[8], lrow[8];
#pragma unroll
  for (int i = 0; i < 8; ++i) { mrow[i] = -3.0e38f; lrow[i] = 0.f; }

  issue_kv_copy(Kh, Vh, 0, kbuf[0], vbuf[0], threadIdx.x);

  for (int j = 0; j <= qb; ++j) {
    wait_kv_copy();                                   // tile j resident in LDS
    if (j < qb)                                       // overlap next tile's DMA
      issue_kv_copy(Kh, Vh, (j + 1) * 64, kbuf[(j + 1) & 1], vbuf[(j + 1) & 1],
                    threadIdx.x);

    const unsigned short* kt = kbuf[j & 1];           // [key_local][d]
    const unsigned short* vt = vbuf[j & 1];           // [d][key_local]
    const int kbase = j * 64;

    // ---- scores: S = Q * K^T (scale folded into Q) ----
    v8f sfr[4];
#pragma unroll
    for (int t = 0; t < 4; ++t) {
      const unsigned short* krow = kt + (size_t)(t * 16 + ln) * HD;
      v8f s = (v8f){0.f,0.f,0.f,0.f,0.f,0.f,0.f,0.f};
      s = wmma_bf16(qa[0], *(const bf16x16*)(krow + halfc * 16), s);
      s = wmma_bf16(qa[1], *(const bf16x16*)(krow + 32 + halfc * 16), s);
      sfr[t] = s;
    }
    // ---- causal mask on diagonal tile ----
    if (j == qb) {
#pragma unroll
      for (int t = 0; t < 4; ++t) {
        const int col = kbase + t * 16 + ln;
#pragma unroll
        for (int i = 0; i < 8; ++i) {
          const int row = m0 + i + 8 * halfc;
          if (col > row) sfr[t][i] = -3.0e38f;
        }
      }
    }
    // ---- online softmax: row max (16 lanes of one half share a row) ----
    float mnew[8], alpha[8];
#pragma unroll
    for (int i = 0; i < 8; ++i) {
      float v = fmaxf(fmaxf(sfr[0][i], sfr[1][i]), fmaxf(sfr[2][i], sfr[3][i]));
#pragma unroll
      for (int off = 1; off < 16; off <<= 1) v = fmaxf(v, __shfl_xor(v, off, 32));
      float mn = fmaxf(mrow[i], v);
      mnew[i]  = mn;
      alpha[i] = __expf(mrow[i] - mn);
      mrow[i]  = mn;
    }
    // ---- exp, write P (bf16) to LDS in row-major [16][64] ----
#pragma unroll
    for (int t = 0; t < 4; ++t) {
#pragma unroll
      for (int i = 0; i < 8; ++i) {
        float p = __expf(sfr[t][i] - mnew[i]);
        sfr[t][i] = p;
        pbuf[w][i + 8 * halfc][t * 16 + ln] = f2bf(p);
      }
    }
    // ---- row sum, rescale running state ----
#pragma unroll
    for (int i = 0; i < 8; ++i) {
      float rs = (sfr[0][i] + sfr[1][i]) + (sfr[2][i] + sfr[3][i]);
#pragma unroll
      for (int off = 1; off < 16; off <<= 1) rs += __shfl_xor(rs, off, 32);
      lrow[i] = lrow[i] * alpha[i] + rs;
    }
#pragma unroll
    for (int t = 0; t < 4; ++t)
#pragma unroll
      for (int i = 0; i < 8; ++i) ctx[t][i] *= alpha[i];

    // ---- ctx += P * V  (A = P from LDS, B = V^T tile from LDS) ----
#pragma unroll
    for (int kk = 0; kk < 2; ++kk) {
      bf16x16 pa = load_a16x32(&pbuf[w][ln][0], kk * 32, halfc);
#pragma unroll
      for (int t = 0; t < 4; ++t) {
        const unsigned short* vrow =
            vt + (size_t)(t * 16 + ln) * 64 + kk * 32 + halfc * 16;
        ctx[t] = wmma_bf16(pa, *(const bf16x16*)vrow, ctx[t]);
      }
    }
  }

  // ---- normalize + store ctx bf16 at [s][h*64+d] ----
#pragma unroll
  for (int i = 0; i < 8; ++i) lrow[i] = 1.0f / lrow[i];
#pragma unroll
  for (int t = 0; t < 4; ++t) {
    const int d = h * HD + t * 16 + ln;
#pragma unroll
    for (int i = 0; i < 8; ++i) {
      const int row = m0 + i + 8 * halfc;
      ctxb[(size_t)row * DM + d] = f2bf(ctx[t][i] * lrow[i]);
    }
  }
}

// ---------------- kernel 4: output projection (32x64 per wave) -> f32 out ----------------
__global__ __launch_bounds__(128) void out_proj_kernel(
    const unsigned short* __restrict__ ctxb,
    const unsigned short* __restrict__ Wob,
    float* __restrict__ out) {
  const int wave  = blockIdx.x * 4 + (threadIdx.x >> 5);
  const int lane  = threadIdx.x & 31;
  const int halfc = lane >> 4;
  const int ln    = lane & 15;
  const int mt = wave >> 4;   // 128 m-tiles of 32 rows
  const int nb = wave & 15;   // 16 n-blocks of 64
  const int m0 = mt * 32, n0 = nb * 64;

  v8f acc[2][4];
#pragma unroll
  for (int am = 0; am < 2; ++am)
#pragma unroll
    for (int t = 0; t < 4; ++t)
      acc[am][t] = (v8f){0.f,0.f,0.f,0.f,0.f,0.f,0.f,0.f};

  const unsigned short* arow0 = ctxb + (size_t)(m0 + ln) * DM;
  const unsigned short* arow1 = ctxb + (size_t)(m0 + 16 + ln) * DM;
  const unsigned short* brow[4];
#pragma unroll
  for (int t = 0; t < 4; ++t)
    brow[t] = Wob + (size_t)(n0 + t * 16 + ln) * DM + halfc * 16;

#pragma unroll 2
  for (int k0 = 0; k0 < DM; k0 += 32) {
    bf16x16 a0 = load_a16x32(arow0, k0, halfc);
    bf16x16 a1 = load_a16x32(arow1, k0, halfc);
    bf16x16 b[4];
#pragma unroll
    for (int t = 0; t < 4; ++t) b[t] = *(const bf16x16*)(brow[t] + k0);
#pragma unroll
    for (int t = 0; t < 4; ++t) acc[0][t] = wmma_bf16(a0, b[t], acc[0][t]);
#pragma unroll
    for (int t = 0; t < 4; ++t) acc[1][t] = wmma_bf16(a1, b[t], acc[1][t]);
  }

#pragma unroll
  for (int am = 0; am < 2; ++am)
#pragma unroll
    for (int t = 0; t < 4; ++t)
#pragma unroll
      for (int i = 0; i < 8; ++i)
        out[(size_t)(m0 + am * 16 + i + 8 * halfc) * DM + n0 + t * 16 + ln] =
            acc[am][t][i];
}

// ---------------- host launcher ----------------
extern "C" void kernel_launch(void* const* d_in, const int* in_sizes, int n_in,
                              void* d_out, int out_size, void* d_ws, size_t ws_size,
                              hipStream_t stream) {
  const float* x  = (const float*)d_in[0];
  const float* Wq = (const float*)d_in[1];
  const float* Wk = (const float*)d_in[2];
  const float* Wv = (const float*)d_in[3];
  const float* Wo = (const float*)d_in[4];
  const int* tokpos = (const int*)d_in[5];
  float* out = (float*)d_out;

  char* ws = (char*)d_ws;
  const size_t MB = 1024u * 1024u;
  unsigned short* xb   = (unsigned short*)(ws + 0 * MB);   // 8 MiB
  unsigned short* Wqb  = (unsigned short*)(ws + 8 * MB);   // 2 MiB
  unsigned short* Wkb  = (unsigned short*)(ws + 10 * MB);  // 2 MiB
  unsigned short* Wvb  = (unsigned short*)(ws + 12 * MB);  // 2 MiB
  unsigned short* Wob  = (unsigned short*)(ws + 14 * MB);  // 2 MiB
  float*          qT   = (float*)(ws + 16 * MB);           // 16 MiB [D][S]
  float*          kT   = (float*)(ws + 32 * MB);           // 16 MiB [D][S]
  unsigned short* Qb   = (unsigned short*)(ws + 48 * MB);  // 8 MiB [h][s][d]
  unsigned short* Kb   = (unsigned short*)(ws + 56 * MB);  // 8 MiB [h][s][d]
  unsigned short* Vtb  = (unsigned short*)(ws + 64 * MB);  // 8 MiB [h*64+d][s]
  unsigned short* ctxb = (unsigned short*)(ws + 72 * MB);  // 8 MiB [s][D]
  // total workspace: 80 MiB

  const int nX = S_LEN * DM;      // 4194304
  const int nW = DM * DM;         // 1048576
  to_bf16_kernel<<<(nX + 255) / 256, 256, 0, stream>>>(x,  xb,  nX);
  to_bf16_kernel<<<(nW + 255) / 256, 256, 0, stream>>>(Wq, Wqb, nW);
  to_bf16_kernel<<<(nW + 255) / 256, 256, 0, stream>>>(Wk, Wkb, nW);
  to_bf16_kernel<<<(nW + 255) / 256, 256, 0, stream>>>(Wv, Wvb, nW);
  to_bf16_kernel<<<(nW + 255) / 256, 256, 0, stream>>>(Wo, Wob, nW);

  // 128 m-tiles(32 rows) * 48 n-blocks = 6144 waves / 4 per block
  qkv_gemm_kernel<<<1536, 128, 0, stream>>>(xb, Wqb, Wkb, Wvb, qT, kT, Vtb);

  rope_pack_kernel<<<dim3(64, 16, 2), 256, 0, stream>>>(qT, kT, tokpos, Qb, Kb);

  flash_attn_kernel<<<dim3(64, 16), 128, 0, stream>>>(Qb, Kb, Vtb, ctxb);

  // 128 m-tiles(32 rows) * 16 n-blocks = 2048 waves / 4 per block
  out_proj_kernel<<<512, 128, 0, stream>>>(ctxb, Wob, out);
}
